// MinkowskiBackbone_28741921144899
// MI455X (gfx1250) — matrix operand order repaired and checked by reading, not compiled, not run on hardware
//
#include <hip/hip_runtime.h>

// ---------------------------------------------------------------------------
// Types for WMMA (CDNA5 / gfx1250, wave32)
// ---------------------------------------------------------------------------
typedef __attribute__((ext_vector_type(16))) __bf16 v16bf;
typedef __attribute__((ext_vector_type(8)))  float  v8f;

struct Half16 { uint4 lo, hi; };   // 32 bytes == 16 bf16 values

__device__ __forceinline__ unsigned short f2bf(float f) {
  // round-to-nearest-even fp32 -> bf16 raw bits
  unsigned int u = __float_as_uint(f);
  u += 0x7FFFu + ((u >> 16) & 1u);
  return (unsigned short)(u >> 16);
}

// ---------------------------------------------------------------------------
// Small utility kernels
// ---------------------------------------------------------------------------
__global__ void zero_f32_kernel(float* __restrict__ p, long long n) {
  long long i = (long long)blockIdx.x * blockDim.x + threadIdx.x;
  if (i < n) p[i] = 0.0f;
}

__global__ void init_scalars_kernel(int* scal, int n0,
                                    const int* n1, const int* n2, const int* n3) {
  if (threadIdx.x == 0 && blockIdx.x == 0) {
    scal[0] = n0; scal[1] = *n1; scal[2] = *n2; scal[3] = *n3;
  }
}

__global__ void f32_to_bf16_kernel(const float* __restrict__ src,
                                   unsigned short* __restrict__ dst, long long n) {
  long long i = (long long)blockIdx.x * blockDim.x + threadIdx.x;
  if (i < n) dst[i] = f2bf(src[i]);
}

// Repack W [K27, Cin, Cout] fp32 -> bf16 in WMMA-B lane order:
// dst[k][kt][nt][lane*16 + h]  <->  (K = kt*32 + 8*(lane/16) + (h<8 ? h : 8+h), N = nt*16 + lane%16)
__global__ void repack_w_kernel(const float* __restrict__ W, unsigned short* __restrict__ dst,
                                int K27, int Cin, int Cout, int Ktiles, int Ntiles) {
  long long idx = (long long)blockIdx.x * blockDim.x + threadIdx.x;
  long long total = (long long)K27 * Ktiles * Ntiles * 512;
  if (idx >= total) return;
  int e = (int)(idx & 511);
  long long rest = idx >> 9;
  int nt = (int)(rest % Ntiles); rest /= Ntiles;
  int kt = (int)(rest % Ktiles);
  int k  = (int)(rest / Ktiles);
  int lane = e >> 4, h = e & 15;
  int Kidx = kt * 32 + ((lane >> 4) << 3) + (h < 8 ? h : 8 + h);
  int Nidx = nt * 16 + (lane & 15);
  unsigned short v = 0;
  if (Kidx < Cin && Nidx < Cout)
    v = f2bf(W[((long long)k * Cin + Kidx) * Cout + Nidx]);
  dst[idx] = v;
}

// ---------------------------------------------------------------------------
// Sparse conv: per-offset gather -> WMMA bf16 GEMM -> atomic scatter-add.
// Block = 128 threads (4 wave32). grid = (ceil(P/16), 27).
// Gather uses GLOBAL_LOAD_ASYNC_TO_LDS_B128 (ASYNCcnt) when rows are >=16B;
// pad rows are redirected to a zeroed scratch line so the DMA writes zeros.
// ---------------------------------------------------------------------------
__global__ __launch_bounds__(128)
void sconv_wmma_kernel(const unsigned short* __restrict__ feats,   // [n_in, Cin] bf16
                       const unsigned short* __restrict__ Bpk,     // [27][Kt][Nt][512] bf16
                       const int* __restrict__ in_map,             // [27, P]
                       const int* __restrict__ out_map,            // [27, P]
                       float* __restrict__ acc,                    // [n_out, Cout] f32
                       const int* __restrict__ n_in_p,
                       const int* __restrict__ n_out_p,
                       const unsigned short* __restrict__ zerobuf, // >=16B of zeros
                       int P, int Cin, int Cout, int Kpad) {
  const int Ktiles = Kpad >> 5;
  const int Ntiles = (Cout + 15) >> 4;
  const int k  = blockIdx.y;
  const int m0 = blockIdx.x << 4;
  const int tid = threadIdx.x;

  __shared__ __align__(16) unsigned short sA[16 * (256 + 8)];
  __shared__ int sIn[16];
  __shared__ int sOut[16];

  const int rowStride = Kpad + 8;     // +16B stagger to avoid LDS bank conflicts
  const int n_in  = *n_in_p;
  const int n_out = *n_out_p;

  if (tid < 16) {
    int i = m0 + tid;
    int vi = -1, vo = -1;
    if (i < P) {
      int a = in_map[(long long)k * P + i];
      int b = out_map[(long long)k * P + i];
      vi = (a >= 0 && a < n_in)  ? a : -1;   // pad value == n_in  -> zero row
      vo = (b >= 0 && b < n_out) ? b : -1;   // pad value == n_out -> drop
    }
    sIn[tid] = vi;
    sOut[tid] = vo;
  }
  __syncthreads();

  const int kshift = 31 - __builtin_clz(Kpad);   // Kpad is a power of two
  if (Cin >= 8) {
    // Async DMA gather: 16B per lane-issue, memory -> LDS, no VGPR round-trip.
    const int chunksPerRow = Kpad >> 3;          // 8 bf16 per 16B chunk
    const int totalChunks = 16 * chunksPerRow;
    for (int e = tid; e < totalChunks; e += 128) {
      int r  = e >> (kshift - 3);
      int c8 = (e & (chunksPerRow - 1)) << 3;
      int srow = sIn[r];
      const unsigned short* gp =
          (srow >= 0 && c8 + 8 <= Cin) ? (feats + (long long)srow * Cin + c8)
                                       : zerobuf;
      // low 32 bits of a generic pointer to LDS == wave-relative LDS byte offset
      unsigned ldsOff = (unsigned)(size_t)(&sA[r * rowStride + c8]);
      unsigned long long ga = (unsigned long long)(size_t)gp;
      asm volatile("global_load_async_to_lds_b128 %0, %1, off"
                   :: "v"(ldsOff), "v"(ga) : "memory");
    }
    asm volatile("s_wait_asynccnt 0x0" ::: "memory");
  } else {
    // Scalar fallback for Cin=4 (rows smaller than one 16B chunk)
    const int total = 16 << kshift;
    for (int e = tid; e < total; e += 128) {
      int r = e >> kshift;
      int c = e & (Kpad - 1);
      int srow = sIn[r];
      unsigned short v = 0;
      if (srow >= 0 && c < Cin) v = feats[(long long)srow * Cin + c];
      sA[r * rowStride + c] = v;
    }
  }
  __syncthreads();

  const int wave = tid >> 5;
  const int lane = tid & 31;
  const int hi = lane >> 4;
  const int lm = lane & 15;

  for (int nt = wave; nt < Ntiles; nt += 4) {
    v8f c = {0.f, 0.f, 0.f, 0.f, 0.f, 0.f, 0.f, 0.f};
    for (int kt = 0; kt < Ktiles; ++kt) {
      // A fragment: row lm, K chunks {kt*32+8*hi ..} and {kt*32+16+8*hi ..}
      const unsigned short* ap = &sA[lm * rowStride + kt * 32 + (hi << 3)];
      Half16 at;
      at.lo = *(const uint4*)(ap);
      at.hi = *(const uint4*)(ap + 16);
      // B fragment: packed contiguous 32B per lane
      const unsigned short* bp =
          Bpk + ((((size_t)k * Ktiles + kt) * Ntiles + nt) << 9) + (lane << 4);
      Half16 bt;
      bt.lo = *(const uint4*)(bp);
      bt.hi = *(const uint4*)(bp + 8);
      if (kt + 1 < Ktiles)
        __builtin_prefetch(bp + ((size_t)Ntiles << 9), 0, 3);  // next K-tile of B
      v16bf a = __builtin_bit_cast(v16bf, at);
      v16bf b = __builtin_bit_cast(v16bf, bt);
      c = __builtin_amdgcn_wmma_f32_16x16x32_bf16(false, a, false, b,
                                                  (short)0, c, false, false);
    }
    // Scatter-add D tile: lane holds (M = r + 8*hi, N = lm) for r = 0..7
    #pragma unroll
    for (int r = 0; r < 8; ++r) {
      int m = r + (hi << 3);
      int orow = sOut[m];
      int col = (nt << 4) + lm;
      if (orow >= 0 && col < Cout)
        atomicAdd(&acc[(long long)orow * Cout + col], c[r]);
    }
  }
}

// ---------------------------------------------------------------------------
// BN statistics: per-channel sum / sumsq with LDS reduction (C is pow2 <= 256)
// ---------------------------------------------------------------------------
__global__ __launch_bounds__(256)
void bn_stats_kernel(const float* __restrict__ acc, const int* __restrict__ n_p,
                     float* __restrict__ sum, float* __restrict__ sumsq, int C) {
  __shared__ float sS[256];
  __shared__ float sQ[256];
  int tid = threadIdx.x;
  if (tid < C) { sS[tid] = 0.f; sQ[tid] = 0.f; }
  __syncthreads();
  long long n = *n_p;
  long long total = n * C;
  long long stride = (long long)gridDim.x * blockDim.x;
  for (long long i = (long long)blockIdx.x * blockDim.x + tid; i < total; i += stride) {
    float x = acc[i];
    int c = (int)(i & (C - 1));
    atomicAdd(&sS[c], x);
    atomicAdd(&sQ[c], x * x);
  }
  __syncthreads();
  if (tid < C) {
    atomicAdd(&sum[tid], sS[tid]);
    atomicAdd(&sumsq[tid], sQ[tid]);
  }
}

// BN apply + ReLU + convert to bf16, writing into (possibly concatenated) dest.
__global__ void bn_apply_kernel(const float* __restrict__ acc, unsigned short* __restrict__ out,
                                const float* __restrict__ g, const float* __restrict__ b,
                                const float* __restrict__ sum, const float* __restrict__ sumsq,
                                const int* __restrict__ n_p, int C, int outStride, int colOff) {
  long long idx = (long long)blockIdx.x * blockDim.x + threadIdx.x;
  int n = *n_p;
  long long total = (long long)n * C;
  if (idx >= total) return;
  int cshift = 31 - __builtin_clz(C);
  long long row = idx >> cshift;
  int c = (int)(idx & (C - 1));
  float fn = (float)n;
  float mean = sum[c] / fn;
  float var = fmaxf(sumsq[c] / fn - mean * mean, 0.f);
  float y = g[c] * (acc[idx] - mean) * rsqrtf(var + 1e-5f) + b[c];
  y = fmaxf(y, 0.f);
  out[row * outStride + colOff + c] = f2bf(y);
}

// Copy skip-connection features into the right half of a concat buffer.
__global__ void concat_copy_kernel(const unsigned short* __restrict__ src,
                                   unsigned short* __restrict__ dst,
                                   const int* __restrict__ n_p, int C,
                                   int dstStride, int colOff) {
  long long idx = (long long)blockIdx.x * blockDim.x + threadIdx.x;
  int n = *n_p;
  if (idx >= (long long)n * C) return;
  int cshift = 31 - __builtin_clz(C);
  long long row = idx >> cshift;
  int c = (int)(idx & (C - 1));
  dst[row * dstStride + colOff + c] = src[idx];
}

// ---------------------------------------------------------------------------
// Final 1x1 conv: out[N0,20] = d2[N0,64] @ Wf (packed, K=64 -> 2 Ktiles,
// N=20 -> 2 Ntiles padded). Block = 64 threads (2 waves), wave w -> ntile w.
// ---------------------------------------------------------------------------
__global__ __launch_bounds__(64)
void final_gemm_kernel(const unsigned short* __restrict__ A,    // [N,64] bf16
                       const unsigned short* __restrict__ Bpk,  // [2][2][512] bf16
                       float* __restrict__ out, int N) {
  int m0 = blockIdx.x << 4;
  int tid = threadIdx.x;
  int nt = tid >> 5;
  int lane = tid & 31;
  int hi = lane >> 4, lm = lane & 15;
  int row = m0 + lm;

  v8f c = {0.f, 0.f, 0.f, 0.f, 0.f, 0.f, 0.f, 0.f};
  for (int kt = 0; kt < 2; ++kt) {
    Half16 at;
    if (row < N) {
      const unsigned short* ap = A + (long long)row * 64 + kt * 32 + (hi << 3);
      at.lo = *(const uint4*)(ap);
      at.hi = *(const uint4*)(ap + 16);
    } else {
      at.lo = uint4{0u, 0u, 0u, 0u};
      at.hi = uint4{0u, 0u, 0u, 0u};
    }
    const unsigned short* bp = Bpk + ((((size_t)kt << 1) + nt) << 9) + (lane << 4);
    Half16 bt;
    bt.lo = *(const uint4*)(bp);
    bt.hi = *(const uint4*)(bp + 8);
    v16bf a = __builtin_bit_cast(v16bf, at);
    v16bf b = __builtin_bit_cast(v16bf, bt);
    c = __builtin_amdgcn_wmma_f32_16x16x32_bf16(false, a, false, b,
                                                (short)0, c, false, false);
  }
  #pragma unroll
  for (int r = 0; r < 8; ++r) {
    int orow = m0 + r + (hi << 3);
    int col = (nt << 4) + lm;
    if (orow < N && col < 20)
      out[(long long)orow * 20 + col] = c[r];
  }
}

// ---------------------------------------------------------------------------
// Host orchestration
// ---------------------------------------------------------------------------
static inline void h_zero(hipStream_t s, float* p, long long n) {
  zero_f32_kernel<<<dim3((unsigned)((n + 255) / 256)), 256, 0, s>>>(p, n);
}

static inline void h_conv(hipStream_t s, const unsigned short* feats,
                          const unsigned short* Bpk, const int* imap, const int* omap,
                          float* acc, const int* nin, const int* nout,
                          const unsigned short* zerobuf,
                          int P, int Cin, int Cout, int Kpad) {
  dim3 grid((unsigned)((P + 15) / 16), 27);
  sconv_wmma_kernel<<<grid, 128, 0, s>>>(feats, Bpk, imap, omap, acc, nin, nout,
                                         zerobuf, P, Cin, Cout, Kpad);
}

static inline void h_bn(hipStream_t s, const float* acc, unsigned short* outF,
                        const float* g, const float* b, float* stats,
                        const int* np, int C, int outStride, int colOff, long long maxN) {
  h_zero(s, stats, 512);
  bn_stats_kernel<<<512, 256, 0, s>>>(acc, np, stats, stats + 256, C);
  long long total = maxN * C;
  bn_apply_kernel<<<dim3((unsigned)((total + 255) / 256)), 256, 0, s>>>(
      acc, outF, g, b, stats, stats + 256, np, C, outStride, colOff);
}

static inline void h_concat(hipStream_t s, const unsigned short* src, unsigned short* dst,
                            const int* np, int C, int dstStride, int colOff, long long maxN) {
  long long total = maxN * C;
  concat_copy_kernel<<<dim3((unsigned)((total + 255) / 256)), 256, 0, s>>>(
      src, dst, np, C, dstStride, colOff);
}

static inline void h_repack(hipStream_t s, const float* W, unsigned short* dst,
                            int K27, int Cin, int Cout, int Kt, int Nt) {
  long long total = (long long)K27 * Kt * Nt * 512;
  repack_w_kernel<<<dim3((unsigned)((total + 255) / 256)), 256, 0, s>>>(
      W, dst, K27, Cin, Cout, Kt, Nt);
}

extern "C" void kernel_launch(void* const* d_in, const int* in_sizes, int n_in_cnt,
                              void* d_out, int out_size, void* d_ws, size_t ws_size,
                              hipStream_t stream) {
  // ---- inputs (setup_inputs dict order) ----
  const float* feats = (const float*)d_in[0];
  const float* W1 = (const float*)d_in[1];
  const float* W2 = (const float*)d_in[2];
  const float* W3 = (const float*)d_in[3];
  const float* W4 = (const float*)d_in[4];
  const float* D4 = (const float*)d_in[5];
  const float* D3 = (const float*)d_in[6];
  const float* D2 = (const float*)d_in[7];
  const float* Wf = (const float*)d_in[8];
  const float* g1 = (const float*)d_in[9];  const float* b1 = (const float*)d_in[10];
  const float* g2 = (const float*)d_in[11]; const float* b2 = (const float*)d_in[12];
  const float* g3 = (const float*)d_in[13]; const float* b3 = (const float*)d_in[14];
  const float* g4 = (const float*)d_in[15]; const float* b4 = (const float*)d_in[16];
  const float* gd4 = (const float*)d_in[17]; const float* bd4 = (const float*)d_in[18];
  const float* gd3 = (const float*)d_in[19]; const float* bd3 = (const float*)d_in[20];
  const float* gd2 = (const float*)d_in[21]; const float* bd2 = (const float*)d_in[22];
  const int* m1i = (const int*)d_in[23]; const int* m1o = (const int*)d_in[24];
  const int* m2i = (const int*)d_in[25]; const int* m2o = (const int*)d_in[26];
  const int* m3i = (const int*)d_in[27]; const int* m3o = (const int*)d_in[28];
  const int* m4i = (const int*)d_in[29]; const int* m4o = (const int*)d_in[30];
  const int* n1p = (const int*)d_in[31];
  const int* n2p = (const int*)d_in[32];
  const int* n3p = (const int*)d_in[33];

  const int N0 = in_sizes[0] / 4;
  const int P1 = in_sizes[23] / 27;
  const int P2 = in_sizes[25] / 27;
  const int P3 = in_sizes[27] / 27;
  const int P4 = in_sizes[29] / 27;
  // hard bounds from the 128^3 grid geometry (cannot read device n1/n2/n3 here)
  const long long N1B = N0;
  const long long N2B = (N0 < 65536) ? N0 : 65536;   // 2 * 32^3
  const long long N3B = (N0 < 8192)  ? N0 : 8192;    // 2 * 16^3

  // ---- workspace carve ----
  char* w = (char*)d_ws;
  auto alloc = [&](size_t bytes) -> void* {
    void* p = (void*)w;
    w += (bytes + 255) & ~(size_t)255;
    return p;
  };
  int*   scal  = (int*)alloc(4 * sizeof(int));     // n0, n1, n2, n3 (device-resident)
  float* stats = (float*)alloc(512 * sizeof(float));
  unsigned short* zbuf = (unsigned short*)alloc(256);  // zeroed line for pad-row DMA

  unsigned short* pW1 = (unsigned short*)alloc((size_t)27 * 1 * 2  * 512 * 2);
  unsigned short* pW2 = (unsigned short*)alloc((size_t)27 * 1 * 4  * 512 * 2);
  unsigned short* pW3 = (unsigned short*)alloc((size_t)27 * 2 * 8  * 512 * 2);
  unsigned short* pW4 = (unsigned short*)alloc((size_t)27 * 4 * 16 * 512 * 2);
  unsigned short* pD4 = (unsigned short*)alloc((size_t)27 * 8 * 8  * 512 * 2);
  unsigned short* pD3 = (unsigned short*)alloc((size_t)27 * 8 * 4  * 512 * 2);
  unsigned short* pD2 = (unsigned short*)alloc((size_t)27 * 4 * 2  * 512 * 2);
  unsigned short* pWf = (unsigned short*)alloc((size_t)1  * 2 * 2  * 512 * 2);

  unsigned short* f0  = (unsigned short*)alloc((size_t)N0 * 4 * 2);
  unsigned short* e1  = (unsigned short*)alloc((size_t)N0 * 32 * 2);
  unsigned short* e2  = (unsigned short*)alloc((size_t)N1B * 64 * 2);
  unsigned short* e3  = (unsigned short*)alloc((size_t)N2B * 128 * 2);
  unsigned short* e4  = (unsigned short*)alloc((size_t)N3B * 256 * 2);
  unsigned short* d4b = (unsigned short*)alloc((size_t)N2B * 256 * 2);
  unsigned short* d3b = (unsigned short*)alloc((size_t)N1B * 128 * 2);
  unsigned short* d2b = (unsigned short*)alloc((size_t)N0 * 64 * 2);

  long long accElems = (long long)N0 * 64;
  if ((long long)N2B * 128 > accElems) accElems = (long long)N2B * 128;
  if ((long long)N3B * 256 > accElems) accElems = (long long)N3B * 256;
  float* acc = (float*)alloc((size_t)accElems * sizeof(float));
  (void)ws_size; (void)n_in_cnt; (void)out_size;

  // ---- pipeline ----
  init_scalars_kernel<<<1, 32, 0, stream>>>(scal, N0, n1p, n2p, n3p);
  h_zero(stream, (float*)zbuf, 64);
  f32_to_bf16_kernel<<<dim3((unsigned)(((long long)N0 * 4 + 255) / 256)), 256, 0, stream>>>(
      feats, f0, (long long)N0 * 4);

  h_repack(stream, W1, pW1, 27, 4,   32,  1, 2);
  h_repack(stream, W2, pW2, 27, 32,  64,  1, 4);
  h_repack(stream, W3, pW3, 27, 64,  128, 2, 8);
  h_repack(stream, W4, pW4, 27, 128, 256, 4, 16);
  h_repack(stream, D4, pD4, 27, 256, 128, 8, 8);
  h_repack(stream, D3, pD3, 27, 256, 64,  8, 4);
  h_repack(stream, D2, pD2, 27, 128, 32,  4, 2);
  h_repack(stream, Wf, pWf, 1,  64,  20,  2, 2);

  // Encoder L1: [N0,4] -> [N0,32]
  h_zero(stream, acc, (long long)N0 * 32);
  h_conv(stream, f0, pW1, m1i, m1o, acc, scal + 0, scal + 0, zbuf, P1, 4, 32, 32);
  h_bn(stream, acc, e1, g1, b1, stats, scal + 0, 32, 32, 0, N0);
  h_concat(stream, e1, d2b, scal + 0, 32, 64, 32, N0);   // skip -> d2[:,32:64]

  // Encoder L2: [N0,32] -> [N1,64]
  h_zero(stream, acc, (long long)N0 * 64);
  h_conv(stream, e1, pW2, m2i, m2o, acc, scal + 0, scal + 1, zbuf, P2, 32, 64, 32);
  h_bn(stream, acc, e2, g2, b2, stats, scal + 1, 64, 64, 0, N1B);
  h_concat(stream, e2, d3b, scal + 1, 64, 128, 64, N1B); // skip -> d3[:,64:128]

  // Encoder L3: [N1,64] -> [N2,128]
  h_zero(stream, acc, (long long)N2B * 128);
  h_conv(stream, e2, pW3, m3i, m3o, acc, scal + 1, scal + 2, zbuf, P3, 64, 128, 64);
  h_bn(stream, acc, e3, g3, b3, stats, scal + 2, 128, 128, 0, N2B);
  h_concat(stream, e3, d4b, scal + 2, 128, 256, 128, N2B); // skip -> d4[:,128:256]

  // Encoder L4: [N2,128] -> [N3,256]
  h_zero(stream, acc, (long long)N3B * 256);
  h_conv(stream, e3, pW4, m4i, m4o, acc, scal + 2, scal + 3, zbuf, P4, 128, 256, 128);
  h_bn(stream, acc, e4, g4, b4, stats, scal + 3, 256, 256, 0, N3B);

  // Decoder D4 (transpose conv: swap maps): [N3,256] -> [N2,128], concat e3
  h_zero(stream, acc, (long long)N2B * 128);
  h_conv(stream, e4, pD4, m4o, m4i, acc, scal + 3, scal + 2, zbuf, P4, 256, 128, 256);
  h_bn(stream, acc, d4b, gd4, bd4, stats, scal + 2, 128, 256, 0, N2B);

  // Decoder D3: [N2,256] -> [N1,64], concat e2
  h_zero(stream, acc, (long long)N1B * 64);
  h_conv(stream, d4b, pD3, m3o, m3i, acc, scal + 2, scal + 1, zbuf, P3, 256, 64, 256);
  h_bn(stream, acc, d3b, gd3, bd3, stats, scal + 1, 64, 128, 0, N1B);

  // Decoder D2: [N1,128] -> [N0,32], concat e1
  h_zero(stream, acc, (long long)N0 * 32);
  h_conv(stream, d3b, pD2, m2o, m2i, acc, scal + 1, scal + 0, zbuf, P2, 128, 32, 128);
  h_bn(stream, acc, d2b, gd2, bd2, stats, scal + 0, 32, 64, 0, N0);

  // Final 1x1: [N0,64] @ [64,20] -> d_out [N0,20] fp32
  final_gemm_kernel<<<dim3((unsigned)((N0 + 15) / 16)), 64, 0, stream>>>(
      d2b, pWf, (float*)d_out, N0);
}
